// EGNN_88476326297727
// MI455X (gfx1250) — matrix operand order repaired and checked by reading, not compiled, not run on hardware
//
#include <hip/hip_runtime.h>
#include <math.h>

#define DEVI __device__ __forceinline__

typedef __attribute__((ext_vector_type(16))) __bf16 v16bf;
typedef __attribute__((ext_vector_type(8)))  __bf16 v8bf;
typedef __attribute__((ext_vector_type(2)))  __bf16 v2bf;
typedef __attribute__((ext_vector_type(8)))  float  v8f;

namespace {
constexpr int NN = 50000;        // nodes
constexpr int NE = 800000;       // edges
constexpr int HD = 64;           // hidden
constexpr int NL = 4;            // layers
constexpr int KE = 193;          // edge MLP1 input dim (2H+1+H)
constexpr int KEPAD = 224;       // 7 * 32
constexpr int KN = 128;          // node MLP1 input dim (2H)
constexpr int ETILES = NE / 16;  // 50000
constexpr int NTILES = NN / 16;  // 3125
constexpr int WPB = 4;           // waves per block in WMMA kernels
}

// ---------------------------------------------------------------- helpers ----

DEVI float fast_rcp(float x) {
#if __has_builtin(__builtin_amdgcn_rcpf)
  return __builtin_amdgcn_rcpf(x);
#else
  return 1.0f / x;
#endif
}
DEVI float fast_sqrt(float x) {
#if __has_builtin(__builtin_amdgcn_sqrtf)
  return __builtin_amdgcn_sqrtf(x);
#else
  return sqrtf(x);
#endif
}
DEVI float siluf(float x) { return x * fast_rcp(1.0f + __expf(-x)); }

DEVI void wave_lds_fence() {
  // wave-private LDS tile: DS ops are in-order per wave; only need the CDNA5
  // split-counter wait before cross-lane reads. "memory" stops reordering.
  asm volatile("s_wait_dscnt 0" ::: "memory");
}

// Async global->LDS copy (gfx1250 GLOBAL_LOAD_ASYNC_TO_LDS_B128, ASYNCcnt).
DEVI void async_copy_16(const void* g, void* l) {
  unsigned long long ga = (unsigned long long)(uintptr_t)g;
  unsigned la = (unsigned)(uintptr_t)l;  // low 32 bits of LDS flat addr = offset
  asm volatile("global_load_async_to_lds_b128 %0, %1, off"
               :: "v"(la), "v"(ga) : "memory");
}
DEVI void async_copy_region(const void* g, void* l, int bytes, int tid, int nthr) {
  for (int o = tid * 16; o < bytes; o += nthr * 16)
    async_copy_16((const char*)g + o, (char*)l + o);
}
DEVI void async_drain() { asm volatile("s_wait_asynccnt 0" ::: "memory"); }

DEVI v8f v8f_zero() {
  v8f z = {0.f, 0.f, 0.f, 0.f, 0.f, 0.f, 0.f, 0.f};
  return z;
}

DEVI void atomic_add_f32(float* p, float v) {
  __hip_atomic_fetch_add(p, v, __ATOMIC_RELAXED, __HIP_MEMORY_SCOPE_AGENT);
}

// A fragment (16x32 bf16, ISA 7.12.2): per lane two contiguous 16B runs.
DEVI v16bf load_a_bf(const __bf16* A, int ld, int kc, int lane) {
  int m = lane & 15, kh = lane >> 4;
  const __bf16* Ar = A + m * ld + kc + kh * 8;
  v8bf lo = *(const v8bf*)Ar;         // K = kc+kh*8 .. +7
  v8bf hi = *(const v8bf*)(Ar + 16);  // K = kc+16+kh*8 .. +7
  return __builtin_shufflevector(lo, hi, 0, 1, 2, 3, 4, 5, 6, 7,
                                 8, 9, 10, 11, 12, 13, 14, 15);
}

DEVI v16bf load_a_f32(const float* A, int ld, int kc, int lane) {
  int m = lane & 15, kh = lane >> 4;
  const float* Ar = A + m * ld + kc + kh * 8;
  v16bf a;
#pragma unroll
  for (int j = 0; j < 8; ++j) {
    a[j] = (__bf16)Ar[j];
    a[8 + j] = (__bf16)Ar[16 + j];
  }
  return a;
}

// B fragment from pre-packed weights: one contiguous 32B run per lane.
// Packed layout: [chunk c][ntile][lane][16 halfwords].
DEVI v16bf load_bp(const __bf16* Wp, int c, int nt, int lane) {
  return *(const v16bf*)(Wp + (size_t)(((c * 4 + nt) * 32 + lane)) * 16);
}

DEVI v8f wmma_bf16(v16bf a, v16bf b, v8f c) {
  return __builtin_amdgcn_wmma_f32_16x16x32_bf16(
      false, a, false, b, (short)0, c, false, false);
}

// ---------------------------------------------------------- small kernels ----

// Pack f32 weight [L][Ksrc][64] into bf16 WMMA B-fragment order
// [L][Kpad/32][4][32][16]. mode 1 = eW1 row permutation (h|h|e|radial).
__global__ void k_pack_b(const float* __restrict__ src, __bf16* __restrict__ dst,
                         int Ksrc, int Kpad, int mode, int total) {
  int i = blockIdx.x * blockDim.x + threadIdx.x;
  if (i >= total) return;
  int j = i & 15;
  int lane = (i >> 4) & 31;
  int nt = (i >> 9) & 3;
  int rest = i >> 11;
  int nchunks = Kpad >> 5;
  int c = rest % nchunks;
  int l = rest / nchunks;
  int v = j >> 1, t = j & 1;
  int kh = lane >> 4, n = lane & 15;
  int k = c * 32 + (j & 7) + kh * 8 + (j >> 3) * 16;
  int col = nt * 16 + n;
  int ks;
  if (mode == 1)
    ks = (k < 128) ? k : (k < 192 ? k + 1 : (k == 192 ? 128 : -1));
  else
    ks = (k < Ksrc) ? k : -1;
  float val = (ks >= 0 && ks < Ksrc) ? src[((size_t)l * Ksrc + ks) * 64 + col] : 0.f;
  dst[i] = (__bf16)val;
}

// plain f32 -> bf16 (for cW2 vector)
__global__ void k_cvt_bf16(const float* __restrict__ src, __bf16* __restrict__ dst,
                           int total) {
  int i = blockIdx.x * blockDim.x + threadIdx.x;
  if (i < total) dst[i] = (__bf16)src[i];
}

__global__ void k_embed_node(const float* __restrict__ hin,
                             const float* __restrict__ W,  // [16,64]
                             const float* __restrict__ b,
                             float* __restrict__ h) {
  int i = blockIdx.x * blockDim.x + threadIdx.x;
  if (i >= NN * HD) return;
  int n = i / HD, j = i % HD;
  float acc = b[j];
#pragma unroll
  for (int k = 0; k < 16; ++k) acc += hin[n * 16 + k] * W[k * HD + j];
  h[i] = acc;
}

__global__ void k_embed_edge(const float* __restrict__ ea,
                             const float* __restrict__ W,  // [8,64]
                             const float* __restrict__ b,
                             float* __restrict__ e) {
  int i = blockIdx.x * blockDim.x + threadIdx.x;
  if (i >= NE * HD) return;
  int ei = i / HD, j = i % HD;
  float acc = b[j];
#pragma unroll
  for (int k = 0; k < 8; ++k) acc += ea[ei * 8 + k] * W[k * HD + j];
  e[i] = acc;
}

__global__ void k_copy_x(const float* __restrict__ xin, float* __restrict__ xbuf) {
  int i = blockIdx.x * blockDim.x + threadIdx.x;
  if (i < NN * 3) xbuf[i] = xin[i];
}

__global__ void k_coord(const float* __restrict__ x,
                        const int* __restrict__ row, const int* __restrict__ col,
                        float* __restrict__ radial, float* __restrict__ cdiff) {
  int ei = blockIdx.x * blockDim.x + threadIdx.x;
  if (ei >= NE) return;
  int r = row[ei], c = col[ei];
  float dx = x[r * 3 + 0] - x[c * 3 + 0];
  float dy = x[r * 3 + 1] - x[c * 3 + 1];
  float dz = x[r * 3 + 2] - x[c * 3 + 2];
  float r2 = dx * dx + dy * dy + dz * dz;
  radial[ei] = r2;
  float inv = fast_rcp(fast_sqrt(r2) + 1.0f);
  cdiff[ei * 3 + 0] = dx * inv;
  cdiff[ei * 3 + 1] = dy * inv;
  cdiff[ei * 3 + 2] = dz * inv;
}

__global__ void k_init_accums(const float* __restrict__ xbuf,
                              float* __restrict__ agg, float* __restrict__ xacc) {
  int i = blockIdx.x * blockDim.x + threadIdx.x;
  if (i >= NN * HD) return;
  agg[i] = 0.0f;
  if (i < NN * 3) xacc[i] = xbuf[i];
}

__global__ void k_finish_x(const float* __restrict__ xacc,
                           const float* __restrict__ nmask,
                           float* __restrict__ xbuf) {
  int i = blockIdx.x * blockDim.x + threadIdx.x;
  if (i >= NN * 3) return;
  xbuf[i] = xacc[i] * nmask[i / 3];
}

__global__ void k_out_h(const float* __restrict__ h,
                        const float* __restrict__ W,  // [64,16]
                        const float* __restrict__ b,
                        const float* __restrict__ nmask,
                        float* __restrict__ out) {
  int i = blockIdx.x * blockDim.x + threadIdx.x;
  if (i >= NN * 16) return;
  int n = i / 16, j = i % 16;
  float acc = b[j];
#pragma unroll 16
  for (int k = 0; k < HD; ++k) acc += h[n * HD + k] * W[k * 16 + j];
  out[i] = acc * nmask[n];
}

__global__ void k_out_x(const float* __restrict__ xbuf, float* __restrict__ out) {
  int i = blockIdx.x * blockDim.x + threadIdx.x;
  if (i < NN * 3) out[i] = xbuf[i];
}

__global__ void k_out_e(const float* __restrict__ e,
                        const float* __restrict__ W,  // [64,8]
                        const float* __restrict__ b,
                        const float* __restrict__ emask,
                        float* __restrict__ out) {
  int i = blockIdx.x * blockDim.x + threadIdx.x;
  if (i >= NE * 8) return;
  int ei = i / 8, j = i % 8;
  float acc = b[j];
#pragma unroll 16
  for (int k = 0; k < HD; ++k) acc += e[(size_t)ei * HD + k] * W[k * 8 + j];
  out[i] = acc * emask[ei];
}

// ------------------------------------------------------- edge MLP (WMMA) ----
// One wave per 16-edge tile. A columns: [h[row](0..63) | h[col](64..127) |
// e(128..191) | radial(192) | pad] — eW1 rows permuted to match at pack time.
__global__ void k_edge_mlp(const float* __restrict__ h,
                           float* __restrict__ e,           // [E,64] in/out
                           const float* __restrict__ radial,
                           const float* __restrict__ cdiff,
                           const float* __restrict__ emask,
                           const int* __restrict__ row,
                           const int* __restrict__ colv,
                           const __bf16* __restrict__ eW1p,  // packed [7][4][32][16]
                           const __bf16* __restrict__ eW2p,  // packed [2][4][32][16]
                           const __bf16* __restrict__ cW1p,  // packed [2][4][32][16]
                           const __bf16* __restrict__ cW2bf, // [64]
                           const float* __restrict__ eb1,
                           const float* __restrict__ eb2,
                           const float* __restrict__ cb1,
                           float* __restrict__ agg,          // [N,64]
                           float* __restrict__ xacc) {       // [N,3]
  extern __shared__ char smem[];
  __bf16* sW1  = (__bf16*)smem;          // 224*64 packed
  __bf16* sW2  = sW1 + KEPAD * HD;       // 64*64 packed
  __bf16* sC1  = sW2 + HD * HD;          // 64*64 packed
  __bf16* sC2v = sC1 + HD * HD;          // 64
  float*  sb   = (float*)(sC2v + HD);    // eb1|eb2|cb1 = 192 floats
  char*   pw   = (char*)(sb + 192);
  int wave = threadIdx.x >> 5;
  int lane = threadIdx.x & 31;
  __bf16* sA = (__bf16*)(pw + wave * (16 * KEPAD * 2 + 16 * HD * 4));
  float*  sM = (float*)(sA + 16 * KEPAD);  // [16][64]

  // async weight tiles into LDS (ASYNCcnt), scalars via normal path
  int tid = threadIdx.x, nthr = blockDim.x;
  async_copy_region(eW1p, sW1, KEPAD * HD * 2, tid, nthr);
  async_copy_region(eW2p, sW2, HD * HD * 2, tid, nthr);
  async_copy_region(cW1p, sC1, HD * HD * 2, tid, nthr);
  for (int i = tid; i < HD; i += nthr) {
    sC2v[i] = cW2bf[i];
    sb[i] = eb1[i];
    sb[64 + i] = eb2[i];
    sb[128 + i] = cb1[i];
  }
  async_drain();
  __syncthreads();

  int tile = blockIdx.x * (blockDim.x >> 5) + wave;
  if (tile >= ETILES) return;
  int eb0 = tile * 16;

  // ---- stage A (bf16, float2 loads -> packed v2bf stores) ----
  for (int r = 0; r < 16; ++r) {
    int ei = eb0 + r;
    int rw = row[ei], cl = colv[ei];
    const float* hr = h + (size_t)rw * HD;
    const float* hc = h + (size_t)cl * HD;
    const float* ev = e + (size_t)ei * HD;
    __bf16* Ar = sA + r * KEPAD;
    float2 a0 = *(const float2*)(hr + 2 * lane);
    float2 a1 = *(const float2*)(hc + 2 * lane);
    float2 a2 = *(const float2*)(ev + 2 * lane);
    v2bf p0; p0[0] = (__bf16)a0.x; p0[1] = (__bf16)a0.y;
    v2bf p1; p1[0] = (__bf16)a1.x; p1[1] = (__bf16)a1.y;
    v2bf p2; p2[0] = (__bf16)a2.x; p2[1] = (__bf16)a2.y;
    *(v2bf*)(Ar + 2 * lane) = p0;
    *(v2bf*)(Ar + 64 + 2 * lane) = p1;
    *(v2bf*)(Ar + 128 + 2 * lane) = p2;
    if (lane == 0) Ar[192] = (__bf16)radial[ei];
    if (lane < 31) Ar[193 + lane] = (__bf16)0.f;  // K pad
  }
  wave_lds_fence();

  int n = lane & 15, mh = lane >> 4;

  // ---- MLP1: [16,224] @ [224,64] ----
  v8f c[4];
#pragma unroll
  for (int nt = 0; nt < 4; ++nt) c[nt] = v8f_zero();
  for (int kc = 0; kc < KEPAD; kc += 32) {
    v16bf a = load_a_bf(sA, KEPAD, kc, lane);
#pragma unroll
    for (int nt = 0; nt < 4; ++nt)
      c[nt] = wmma_bf16(a, load_bp(sW1, kc >> 5, nt, lane), c[nt]);
  }
#pragma unroll
  for (int nt = 0; nt < 4; ++nt) {
    float bias = sb[nt * 16 + n];
#pragma unroll
    for (int v = 0; v < 8; ++v) {
      int m = v + mh * 8;
      sM[m * HD + nt * 16 + n] = siluf(c[nt][v] + bias);
    }
  }
  // stage edge mask tile (sA tail is free now)
  float* smask = (float*)sA + 1536;
  if (lane < 16) smask[lane] = emask[eb0 + lane];
  wave_lds_fence();

  // ---- MLP2: [16,64] @ [64,64], silu * edge_mask -> m ----
  v8f d[4];
#pragma unroll
  for (int nt = 0; nt < 4; ++nt) d[nt] = v8f_zero();
  for (int kc = 0; kc < HD; kc += 32) {
    v16bf a = load_a_f32(sM, HD, kc, lane);
#pragma unroll
    for (int nt = 0; nt < 4; ++nt)
      d[nt] = wmma_bf16(a, load_bp(sW2, kc >> 5, nt, lane), d[nt]);
  }
#pragma unroll
  for (int nt = 0; nt < 4; ++nt) {
    float bias = sb[64 + nt * 16 + n];
#pragma unroll
    for (int v = 0; v < 8; ++v) {
      int m = v + mh * 8;
      sM[m * HD + nt * 16 + n] = siluf(d[nt][v] + bias) * smask[m];
    }
  }
  wave_lds_fence();

  // ---- write updated edge features + scatter-add messages ----
  for (int r = 0; r < 16; ++r) {
    size_t eo = (size_t)(eb0 + r) * HD;
    e[eo + lane]      = sM[r * HD + lane];
    e[eo + 32 + lane] = sM[r * HD + 32 + lane];
    int rw = row[eb0 + r];
    atomic_add_f32(&agg[(size_t)rw * HD + lane],      sM[r * HD + lane]);
    atomic_add_f32(&agg[(size_t)rw * HD + 32 + lane], sM[r * HD + 32 + lane]);
  }

  // ---- coord head: q = silu(m @ cW1 + cb1); p = q @ cW2 ----
  v16bf af0 = load_a_f32(sM, HD, 0, lane);
  v16bf af1 = load_a_f32(sM, HD, 32, lane);
  v8f q[4];
#pragma unroll
  for (int nt = 0; nt < 4; ++nt) {
    q[nt] = v8f_zero();
    q[nt] = wmma_bf16(af0, load_bp(sC1, 0, nt, lane), q[nt]);
    q[nt] = wmma_bf16(af1, load_bp(sC1, 1, nt, lane), q[nt]);
  }
  float* sQ = (float*)sA;  // [16][64] scratch (below smask)
#pragma unroll
  for (int nt = 0; nt < 4; ++nt) {
    float bias = sb[128 + nt * 16 + n];
#pragma unroll
    for (int v = 0; v < 8; ++v) {
      int m = v + mh * 8;
      sQ[m * HD + nt * 16 + n] = siluf(q[nt][v] + bias);
    }
  }
  wave_lds_fence();
  if (lane < 16) {
    float p = 0.f;
#pragma unroll 8
    for (int k = 0; k < HD; ++k) p += sQ[lane * HD + k] * (float)sC2v[k];
    int ei = eb0 + lane;
    int rw = row[ei];
    atomic_add_f32(&xacc[rw * 3 + 0], cdiff[ei * 3 + 0] * p);
    atomic_add_f32(&xacc[rw * 3 + 1], cdiff[ei * 3 + 1] * p);
    atomic_add_f32(&xacc[rw * 3 + 2], cdiff[ei * 3 + 2] * p);
  }
}

// ------------------------------------------------------- node MLP (WMMA) ----
__global__ void k_node_mlp(float* __restrict__ h,
                           const float* __restrict__ agg,
                           const float* __restrict__ nmask,
                           const __bf16* __restrict__ nW1p,  // packed [4][4][32][16]
                           const __bf16* __restrict__ nW2p,  // packed [2][4][32][16]
                           const float* __restrict__ nb1,
                           const float* __restrict__ nb2) {
  extern __shared__ char smem[];
  __bf16* sW1 = (__bf16*)smem;            // 128*64 packed
  __bf16* sW2 = sW1 + KN * HD;            // 64*64 packed
  float*  sb  = (float*)(sW2 + HD * HD);  // nb1|nb2 = 128 floats
  char*   pw  = (char*)(sb + 128);
  int wave = threadIdx.x >> 5;
  int lane = threadIdx.x & 31;
  __bf16* sA = (__bf16*)(pw + wave * (16 * KN * 2 + 16 * HD * 4));
  float*  sM = (float*)(sA + 16 * KN);

  int tid = threadIdx.x, nthr = blockDim.x;
  async_copy_region(nW1p, sW1, KN * HD * 2, tid, nthr);
  async_copy_region(nW2p, sW2, HD * HD * 2, tid, nthr);
  for (int i = tid; i < HD; i += nthr) {
    sb[i] = nb1[i];
    sb[64 + i] = nb2[i];
  }
  async_drain();
  __syncthreads();

  int tile = blockIdx.x * (blockDim.x >> 5) + wave;
  if (tile >= NTILES) return;
  int nb0 = tile * 16;

  for (int r = 0; r < 16; ++r) {
    const float* hr = h + (size_t)(nb0 + r) * HD;
    const float* ar = agg + (size_t)(nb0 + r) * HD;
    __bf16* Ar = sA + r * KN;
    float2 a0 = *(const float2*)(hr + 2 * lane);
    float2 a1 = *(const float2*)(ar + 2 * lane);
    v2bf p0; p0[0] = (__bf16)a0.x; p0[1] = (__bf16)a0.y;
    v2bf p1; p1[0] = (__bf16)a1.x; p1[1] = (__bf16)a1.y;
    *(v2bf*)(Ar + 2 * lane) = p0;
    *(v2bf*)(Ar + 64 + 2 * lane) = p1;
  }
  wave_lds_fence();

  int n = lane & 15, mh = lane >> 4;

  v8f c[4];
#pragma unroll
  for (int nt = 0; nt < 4; ++nt) c[nt] = v8f_zero();
  for (int kc = 0; kc < KN; kc += 32) {
    v16bf a = load_a_bf(sA, KN, kc, lane);
#pragma unroll
    for (int nt = 0; nt < 4; ++nt)
      c[nt] = wmma_bf16(a, load_bp(sW1, kc >> 5, nt, lane), c[nt]);
  }
#pragma unroll
  for (int nt = 0; nt < 4; ++nt) {
    float bias = sb[nt * 16 + n];
#pragma unroll
    for (int v = 0; v < 8; ++v) {
      int m = v + mh * 8;
      sM[m * HD + nt * 16 + n] = siluf(c[nt][v] + bias);
    }
  }
  wave_lds_fence();

  v8f d[4];
#pragma unroll
  for (int nt = 0; nt < 4; ++nt) d[nt] = v8f_zero();
  for (int kc = 0; kc < HD; kc += 32) {
    v16bf a = load_a_f32(sM, HD, kc, lane);
#pragma unroll
    for (int nt = 0; nt < 4; ++nt)
      d[nt] = wmma_bf16(a, load_bp(sW2, kc >> 5, nt, lane), d[nt]);
  }
#pragma unroll
  for (int nt = 0; nt < 4; ++nt) {
    float bias = sb[64 + nt * 16 + n];
#pragma unroll
    for (int v = 0; v < 8; ++v) {
      int m = v + mh * 8;
      int node = nb0 + m;
      size_t idx = (size_t)node * HD + nt * 16 + n;
      float out = d[nt][v] + bias;
      h[idx] = (h[idx] + out) * nmask[node];
    }
  }
}

// -------------------------------------------------------------- launcher ----

extern "C" void kernel_launch(void* const* d_in, const int* in_sizes, int n_in,
                              void* d_out, int out_size, void* d_ws, size_t ws_size,
                              hipStream_t stream) {
  const float* h_in   = (const float*)d_in[0];
  const float* x_in   = (const float*)d_in[1];
  const float* ea_in  = (const float*)d_in[2];
  const float* nmask  = (const float*)d_in[3];
  const float* emask  = (const float*)d_in[4];
  const float* Wn_in  = (const float*)d_in[5];
  const float* bn_in  = (const float*)d_in[6];
  const float* Wn_out = (const float*)d_in[7];
  const float* bn_out = (const float*)d_in[8];
  const float* We_in  = (const float*)d_in[9];
  const float* be_in  = (const float*)d_in[10];
  const float* We_out = (const float*)d_in[11];
  const float* be_out = (const float*)d_in[12];
  const float* eW1    = (const float*)d_in[13];
  const float* eb1    = (const float*)d_in[14];
  const float* eW2    = (const float*)d_in[15];
  const float* eb2    = (const float*)d_in[16];
  const float* nW1    = (const float*)d_in[17];
  const float* nb1    = (const float*)d_in[18];
  const float* nW2    = (const float*)d_in[19];
  const float* nb2    = (const float*)d_in[20];
  const float* cW1    = (const float*)d_in[21];
  const float* cb1    = (const float*)d_in[22];
  const float* cW2    = (const float*)d_in[23];
  const int*   eidx   = (const int*)d_in[24];
  const int* row = eidx;
  const int* col = eidx + NE;

  char* ws = (char*)d_ws;
  size_t off = 0;
  auto carve = [&](size_t bytes) {
    char* p = ws + off;
    off = (off + bytes + 255) & ~(size_t)255;
    return p;
  };
  float* h      = (float*)carve((size_t)NN * HD * 4);
  float* e      = (float*)carve((size_t)NE * HD * 4);
  float* agg    = (float*)carve((size_t)NN * HD * 4);
  float* xbuf   = (float*)carve((size_t)NN * 3 * 4);
  float* xacc   = (float*)carve((size_t)NN * 3 * 4);
  float* radial = (float*)carve((size_t)NE * 4);
  float* cdiff  = (float*)carve((size_t)NE * 3 * 4);
  __bf16* eW1p  = (__bf16*)carve((size_t)NL * KEPAD * HD * 2);
  __bf16* eW2p  = (__bf16*)carve((size_t)NL * HD * HD * 2);
  __bf16* cW1p  = (__bf16*)carve((size_t)NL * HD * HD * 2);
  __bf16* cW2bf = (__bf16*)carve((size_t)NL * HD * 2);
  __bf16* nW1p  = (__bf16*)carve((size_t)NL * KN * HD * 2);
  __bf16* nW2p  = (__bf16*)carve((size_t)NL * HD * HD * 2);
  (void)ws_size; (void)in_sizes; (void)n_in;

  const int T = 256;
  auto blocks = [](long long cnt, int t) { return (unsigned)((cnt + t - 1) / t); };

  // weight packing (bf16, WMMA fragment order; eW1 row-permuted + K-padded)
  k_pack_b<<<blocks((long long)NL * KEPAD * HD, T), T, 0, stream>>>(eW1, eW1p, KE, KEPAD, 1, NL * KEPAD * HD);
  k_pack_b<<<blocks((long long)NL * HD * HD, T), T, 0, stream>>>(eW2, eW2p, HD, HD, 0, NL * HD * HD);
  k_pack_b<<<blocks((long long)NL * HD * HD, T), T, 0, stream>>>(cW1, cW1p, HD, HD, 0, NL * HD * HD);
  k_pack_b<<<blocks((long long)NL * KN * HD, T), T, 0, stream>>>(nW1, nW1p, KN, KN, 0, NL * KN * HD);
  k_pack_b<<<blocks((long long)NL * HD * HD, T), T, 0, stream>>>(nW2, nW2p, HD, HD, 0, NL * HD * HD);
  k_cvt_bf16<<<blocks((long long)NL * HD, T), T, 0, stream>>>(cW2, cW2bf, NL * HD);

  // embeddings + coord copy
  k_embed_node<<<blocks((long long)NN * HD, T), T, 0, stream>>>(h_in, Wn_in, bn_in, h);
  k_embed_edge<<<blocks((long long)NE * HD, T), T, 0, stream>>>(ea_in, We_in, be_in, e);
  k_copy_x<<<blocks((long long)NN * 3, T), T, 0, stream>>>(x_in, xbuf);

  constexpr size_t SM_EDGE =
      (size_t)KEPAD * HD * 2 + 2 * HD * HD * 2 + HD * 2 + 192 * 4 +
      (size_t)WPB * (16 * KEPAD * 2 + 16 * HD * 4);
  constexpr size_t SM_NODE =
      (size_t)KN * HD * 2 + HD * HD * 2 + 128 * 4 +
      (size_t)WPB * (16 * KN * 2 + 16 * HD * 4);

  for (int l = 0; l < NL; ++l) {
    k_coord<<<blocks(NE, T), T, 0, stream>>>(xbuf, row, col, radial, cdiff);
    k_init_accums<<<blocks((long long)NN * HD, T), T, 0, stream>>>(xbuf, agg, xacc);
    k_edge_mlp<<<ETILES / WPB, WPB * 32, SM_EDGE, stream>>>(
        h, e, radial, cdiff, emask, row, col,
        eW1p + (size_t)l * KEPAD * HD, eW2p + (size_t)l * HD * HD,
        cW1p + (size_t)l * HD * HD, cW2bf + (size_t)l * HD,
        eb1 + l * HD, eb2 + l * HD, cb1 + l * HD, agg, xacc);
    k_node_mlp<<<(NTILES + WPB - 1) / WPB, WPB * 32, SM_NODE, stream>>>(
        h, agg, nmask,
        nW1p + (size_t)l * KN * HD, nW2p + (size_t)l * HD * HD,
        nb1 + l * HD, nb2 + l * HD);
    k_finish_x<<<blocks((long long)NN * 3, T), T, 0, stream>>>(xacc, nmask, xbuf);
  }

  float* out  = (float*)d_out;
  float* outh = out;
  float* outx = out + (size_t)NN * 16;
  float* oute = out + (size_t)NN * 16 + (size_t)NN * 3;
  k_out_h<<<blocks((long long)NN * 16, T), T, 0, stream>>>(h, Wn_out, bn_out, nmask, outh);
  k_out_x<<<blocks((long long)NN * 3, T), T, 0, stream>>>(xbuf, outx);
  k_out_e<<<blocks((long long)NE * 8, T), T, 0, stream>>>(e, We_out, be_out, emask, oute);
}
